// LayoutGraphModel_58969900974117
// MI455X (gfx1250) — compile-verified, matrix operand order checked by reading, not run on hardware
//
#include <hip/hip_runtime.h>
#include <stdint.h>

typedef unsigned short u16;
typedef unsigned int   u32;

typedef __attribute__((ext_vector_type(16))) __bf16 v16bf;
typedef __attribute__((ext_vector_type(8)))  float  v8f;

// ---------- scalar helpers ----------
__device__ __forceinline__ u16 f2bf(float f) {
  union { float f; u32 u; } v; v.f = f;
  u32 r = v.u + 0x7FFFu + ((v.u >> 16) & 1u);   // RNE
  return (u16)(r >> 16);
}
__device__ __forceinline__ float bf2f(u16 h) {
  union { u32 u; float f; } v; v.u = ((u32)h) << 16; return v.f;
}
// order-preserving float -> u32 key (for atomicMax-based segment max)
__device__ __forceinline__ u32 fkey(float f) {
  union { float f; u32 u; } v; v.f = f;
  return (v.u & 0x80000000u) ? ~v.u : (v.u | 0x80000000u);
}
__device__ __forceinline__ float keyf(u32 k) {
  union { u32 u; float f; } v;
  v.u = (k & 0x80000000u) ? (k & 0x7fffffffu) : ~k;
  return v.f;
}

// ---------- WMMA fragment load ----------
// 16-bit A-matrix 16x32 lane layout (ISA 7.12.2): lanes 0-15 hold rows M=0..15
// with K=[0..7],[16..23]; lanes 16-31 hold same rows with K=[8..15],[24..31].
// B fragments use the same pattern on the transposed (column-per-lane) weight.
__device__ __forceinline__ v16bf load_frag(const u16* __restrict__ base, int stride,
                                           int row0, int k0, int lane) {
  int r  = row0 + (lane & 15);
  int kb = k0 + ((lane >> 4) << 3);
  const u16* p = base + (size_t)r * stride + kb;
  union { uint4 u[2]; v16bf v; } un;
  un.u[0] = *(const uint4*)(p);        // K = kb .. kb+7
  un.u[1] = *(const uint4*)(p + 16);   // K = kb+16 .. kb+23
  return un.v;
}

// ---------- GEMM: Out[N,M](bf16) = act(A[N,K](bf16) @ Bt[M,K]^T(bf16) + bias) ----------
// wave: 32 rows x NT*16 cols (2 x NT accumulators, statically indexed);
// block: 4 waves = 128 rows. NT is a compile-time template param so every
// accumulator / fragment index is static -> no M0-indexed VGPR moves.
template<int NT>
__global__ __launch_bounds__(128) void gemm_bf16(
    const u16* __restrict__ A, const u16* __restrict__ Bt,
    const float* __restrict__ bias, u16* __restrict__ Out,
    int K, int ldOut, float alpha) {
  int lane = threadIdx.x & 31;
  int wave = threadIdx.x >> 5;
  int rowBase = blockIdx.x * 128 + wave * 32;
  int colBase = blockIdx.y * (NT * 16);

  v8f z = {0.f,0.f,0.f,0.f,0.f,0.f,0.f,0.f};
  v8f acc[2][NT];
#pragma unroll
  for (int r = 0; r < 2; ++r)
#pragma unroll
    for (int t = 0; t < NT; ++t) acc[r][t] = z;

  for (int k = 0; k < K; k += 32) {
    v16bf a0 = load_frag(A, K, rowBase,      k, lane);
    v16bf a1 = load_frag(A, K, rowBase + 16, k, lane);
    v16bf b[NT];
#pragma unroll
    for (int t = 0; t < NT; ++t) b[t] = load_frag(Bt, K, colBase + t * 16, k, lane);
#pragma unroll
    for (int t = 0; t < NT; ++t) {
      acc[0][t] = __builtin_amdgcn_wmma_f32_16x16x32_bf16(
          false, a0, false, b[t], (short)0, acc[0][t], false, false);
      acc[1][t] = __builtin_amdgcn_wmma_f32_16x16x32_bf16(
          false, a1, false, b[t], (short)0, acc[1][t], false, false);
    }
  }
  // C/D layout: VGPR i, lanes 0-15 -> (M=i, N=lane); lanes 16-31 -> (M=8+i, N=lane-16)
  int col0 = colBase + (lane & 15);
#pragma unroll
  for (int r = 0; r < 2; ++r) {
    int row0 = rowBase + r * 16 + ((lane >> 4) << 3);
#pragma unroll
    for (int t = 0; t < NT; ++t) {
      int col = col0 + t * 16;
      float bb = bias[col];
#pragma unroll
      for (int i = 0; i < 8; ++i) {
        float v = acc[r][t][i] + bb;
        v = v > 0.f ? v : v * alpha;   // alpha==1.0 -> identity
        Out[(size_t)(row0 + i) * ldOut + col] = f2bf(v);
      }
    }
  }
}

// ---------- weight prep: Wt[m, kOff+k] = bf16(W[k, m]) with zero pad ----------
__global__ void wprep(const float* __restrict__ W, u16* __restrict__ Wt,
                      int Kin, int Min, int Kblk, int Mblk, int ldT, int kOff) {
  int idx = blockIdx.x * 256 + threadIdx.x;
  if (idx >= Mblk * Kblk) return;
  int m = idx / Kblk, k = idx % Kblk;
  float v = (k < Kin && m < Min) ? W[(size_t)k * Min + m] : 0.f;
  Wt[(size_t)m * ldT + kOff + k] = f2bf(v);
}
__global__ void bias_pad(const float* __restrict__ b, float* __restrict__ bp,
                         int nin, int npad) {
  int i = blockIdx.x * 256 + threadIdx.x;
  if (i < npad) bp[i] = (i < nin) ? b[i] : 0.f;
}

// ---------- feature build ----------
__global__ void build_xin(const float* __restrict__ nf, const int* __restrict__ ops,
                          const float* __restrict__ emb, u16* __restrict__ xin, size_t total) {
  size_t i = (size_t)blockIdx.x * 256 + threadIdx.x;
  if (i >= total) return;
  int node = (int)(i / 192), c = (int)(i % 192);
  float v;
  if (c < 112)      v = nf[(size_t)node * 112 + c];
  else if (c < 176) v = emb[(size_t)ops[node] * 64 + (c - 112)];
  else              v = 0.f;
  xin[i] = f2bf(v);
}
__global__ void build_cfin(const float* __restrict__ ncf, u16* __restrict__ cfin, size_t total) {
  size_t i = (size_t)blockIdx.x * 256 + threadIdx.x;
  if (i >= total) return;
  int node = (int)(i >> 5), c = (int)(i & 31);
  cfin[i] = f2bf((c < 14) ? ncf[(size_t)node * 14 + c] : 0.f);
}
// out[node*ldOut + colOff + c] = c<64 ? cf[node,c] : h[node, c-64]   (c in 0..255)
__global__ void concat256(const u16* __restrict__ cf, const u16* __restrict__ hx,
                          u16* __restrict__ out, int ldOut, int colOff, size_t total) {
  size_t i = (size_t)blockIdx.x * 256 + threadIdx.x;
  if (i >= total) return;
  int node = (int)(i >> 8), c = (int)(i & 255);
  u16 v = (c < 64) ? cf[(size_t)node * 64 + c] : hx[(size_t)node * 192 + (c - 64)];
  out[(size_t)node * ldOut + colOff + c] = v;
}

// ---------- misc fills ----------
__global__ void fill_u32(u32* __restrict__ p, u32 v, size_t n) {
  size_t i = (size_t)blockIdx.x * 256 + threadIdx.x;
  if (i < n) p[i] = v;
}
__global__ void zero_u16(u16* __restrict__ p, size_t n) {
  size_t i = (size_t)blockIdx.x * 256 + threadIdx.x;
  if (i < n) p[i] = 0;
}

// ---------- edge scatter-max (segment_max over dst) ----------
__global__ void scatter_max(const int* __restrict__ edges, const u16* __restrict__ zbuf,
                            u32* __restrict__ aggK, int E) {
  size_t tid = (size_t)blockIdx.x * 256 + threadIdx.x;
  if (tid >= (size_t)E * 64) return;
  int e  = (int)(tid >> 6);
  int c4 = ((int)tid & 63) << 2;
  int src = edges[e];
  int dst = edges[E + e];
  const u16* yp = zbuf + (size_t)src * 512 + 256 + c4;  // y_in lives in cols 256..511
  uint2 d = *(const uint2*)yp;                           // 4 bf16
  u16 a0 = (u16)(d.x & 0xFFFFu), a1 = (u16)(d.x >> 16);
  u16 a2 = (u16)(d.y & 0xFFFFu), a3 = (u16)(d.y >> 16);
  u32* kp = aggK + (size_t)dst * 256 + c4;
  atomicMax(&kp[0], fkey(bf2f(a0)));
  atomicMax(&kp[1], fkey(bf2f(a1)));
  atomicMax(&kp[2], fkey(bf2f(a2)));
  atomicMax(&kp[3], fkey(bf2f(a3)));
}
__global__ void decode_agg(const u32* __restrict__ aggK, u16* __restrict__ zbuf, size_t total) {
  size_t i = (size_t)blockIdx.x * 256 + threadIdx.x;
  if (i >= total) return;
  int node = (int)(i >> 8), c = (int)(i & 255);
  float f = keyf(aggK[i]);
  if (!(f == f) || fabsf(f) > 3.0e38f) f = 0.f;  // NaN / +-inf (isolated node) -> 0
  zbuf[(size_t)node * 512 + c] = f2bf(f);
}

// ---------- residual ----------
__global__ void residual_add(u16* __restrict__ h, const u16* __restrict__ y, size_t n) {
  size_t i = (size_t)blockIdx.x * 256 + threadIdx.x;
  if (i < n) h[i] = f2bf(bf2f(h[i]) + bf2f(y[i]));
}

// ---------- per-segment sum pool + fc ----------
__global__ void pool_fc(const u16* __restrict__ h, const int* __restrict__ sep,
                        const float* __restrict__ fc_w, float* __restrict__ out) {
  int s = blockIdx.x, c = threadIdx.x;      // 192 threads
  int start = (s == 0) ? 0 : sep[s - 1];
  int end   = sep[s];
  float acc = 0.f;
  for (int n = start; n < end; ++n) acc += bf2f(h[(size_t)n * 192 + c]);
  acc *= fc_w[c];
  __shared__ float red[192];
  red[c] = acc; __syncthreads();
  if (c < 64) red[c] = red[c] + red[c + 64] + red[c + 128];
  __syncthreads();
  for (int o = 32; o > 0; o >>= 1) { if (c < o) red[c] += red[c + o]; __syncthreads(); }
  if (c == 0) out[s] = red[0];
}

// =====================================================================
extern "C" void kernel_launch(void* const* d_in, const int* in_sizes, int n_in,
                              void* d_out, int out_size, void* d_ws, size_t ws_size,
                              hipStream_t stream) {
  const float* nfeat  = (const float*)d_in[0];
  const float* ncf    = (const float*)d_in[1];
  const int*   sep    = (const int*)  d_in[2];
  const int*   ops    = (const int*)  d_in[3];
  const int*   edges  = (const int*)  d_in[4];
  const float* emb    = (const float*)d_in[6];
  const float* nf_w1  = (const float*)d_in[7];
  const float* nf_b1  = (const float*)d_in[8];
  const float* nf_w2  = (const float*)d_in[9];
  const float* nf_b2  = (const float*)d_in[10];
  const float* cf_w1  = (const float*)d_in[11];
  const float* cf_b1  = (const float*)d_in[12];
  const float* cf_w2  = (const float*)d_in[13];
  const float* cf_b2  = (const float*)d_in[14];
  const float* pre_w1 = (const float*)d_in[15];
  const float* pre_b1 = (const float*)d_in[16];
  const float* pre_w2 = (const float*)d_in[17];
  const float* pre_b2 = (const float*)d_in[18];
  const float* sage_wl= (const float*)d_in[19];
  const float* sage_bl= (const float*)d_in[20];
  const float* sage_wr= (const float*)d_in[21];
  const float* mlp_w1 = (const float*)d_in[22];
  const float* mlp_b1 = (const float*)d_in[23];
  const float* mlp_w2 = (const float*)d_in[24];
  const float* mlp_b2 = (const float*)d_in[25];
  const float* fc_w   = (const float*)d_in[26];
  float* out = (float*)d_out;

  const int N    = in_sizes[0] / 112;   // 262144
  const int E    = in_sizes[4] / 2;     // 524288
  const int nseg = in_sizes[2];         // 512

  char* ws = (char*)d_ws;
  size_t off = 0;
  auto alloc = [&](size_t bytes) -> char* {
    off = (off + 255) & ~(size_t)255;
    char* p = ws + off; off += bytes; return p;
  };

  // transposed bf16 weights [M, K]
  u16* nf_w1t  = (u16*)alloc(192 * 192 * 2);
  u16* nf_w2t  = (u16*)alloc(192 * 192 * 2);
  u16* cf_w1t  = (u16*)alloc(16 * 32 * 2);
  u16* cf_w2t  = (u16*)alloc(64 * 32 * 2);
  u16* pre_w1t = (u16*)alloc(192 * 256 * 2);
  u16* pre_w2t = (u16*)alloc(192 * 192 * 2);
  u16* sageT[2]; sageT[0] = (u16*)alloc(256 * 512 * 2); sageT[1] = (u16*)alloc(256 * 512 * 2);
  u16* mlp1T[2]; mlp1T[0] = (u16*)alloc(192 * 256 * 2); mlp1T[1] = (u16*)alloc(192 * 256 * 2);
  u16* mlp2T[2]; mlp2T[0] = (u16*)alloc(192 * 192 * 2); mlp2T[1] = (u16*)alloc(192 * 192 * 2);
  float* nf_b1p = (float*)alloc(192 * 4);
  float* cf_b1p = (float*)alloc(16 * 4);
  // activations
  u16* bufA = (u16*)alloc((size_t)N * 192 * 2);   // x_in -> x -> mlp hidden
  u16* bufB = (u16*)alloc((size_t)N * 192 * 2);   // t1 -> t2 -> y3
  u16* bufH = (u16*)alloc((size_t)N * 192 * 2);   // h (residual stream)
  u16* cfin = (u16*)alloc((size_t)N * 32 * 2);
  u16* cf1  = (u16*)alloc((size_t)N * 32 * 2);
  u16* cfb  = (u16*)alloc((size_t)N * 64 * 2);    // cf
  u16* zbuf = (u16*)alloc((size_t)N * 512 * 2);   // [agg | y_in]
  u32* aggK = (u32*)alloc((size_t)N * 256 * 4);
  u16* bufY = (u16*)alloc((size_t)N * 256 * 2);   // h_in / y1

  auto blocks = [](size_t n) { return (unsigned)((n + 255) / 256); };
  auto wp = [&](const float* W, u16* Wt, int Kin, int Min, int Kblk, int Mblk, int ldT, int kOff) {
    wprep<<<blocks((size_t)Kblk * Mblk), 256, 0, stream>>>(W, Wt, Kin, Min, Kblk, Mblk, ldT, kOff);
  };
  // M is always 16 (NT=1) or a multiple of 64 (NT=4) in this model.
  auto gemm = [&](const u16* A, const u16* Bt, const float* bias, u16* Out,
                  int K, int M, int ldOut, float alpha) {
    if (M == 16) {
      dim3 grid((unsigned)(N / 128), 1);
      gemm_bf16<1><<<grid, 128, 0, stream>>>(A, Bt, bias, Out, K, ldOut, alpha);
    } else {
      dim3 grid((unsigned)(N / 128), (unsigned)(M / 64));
      gemm_bf16<4><<<grid, 128, 0, stream>>>(A, Bt, bias, Out, K, ldOut, alpha);
    }
  };

  // ---- weight / bias prep (re-done every call; deterministic) ----
  wp(nf_w1, nf_w1t, 176, 176, 192, 192, 192, 0);
  wp(nf_w2, nf_w2t, 176, 192, 192, 192, 192, 0);
  wp(cf_w1, cf_w1t, 14, 14, 32, 16, 32, 0);
  wp(cf_w2, cf_w2t, 14, 64, 32, 64, 32, 0);
  wp(pre_w1, pre_w1t, 256, 192, 256, 192, 256, 0);
  wp(pre_w2, pre_w2t, 192, 192, 192, 192, 192, 0);
  for (int l = 0; l < 2; ++l) {
    wp(sage_wl + (size_t)l * 256 * 256, sageT[l], 256, 256, 256, 256, 512, 0);
    wp(sage_wr + (size_t)l * 256 * 256, sageT[l], 256, 256, 256, 256, 512, 256);
    wp(mlp_w1 + (size_t)l * 256 * 192, mlp1T[l], 256, 192, 256, 192, 256, 0);
    wp(mlp_w2 + (size_t)l * 192 * 192, mlp2T[l], 192, 192, 192, 192, 192, 0);
  }
  bias_pad<<<1, 256, 0, stream>>>(nf_b1, nf_b1p, 176, 192);
  bias_pad<<<1, 256, 0, stream>>>(cf_b1, cf_b1p, 14, 16);

  // ---- node feature MLP ----
  build_xin<<<blocks((size_t)N * 192), 256, 0, stream>>>(nfeat, ops, emb, bufA, (size_t)N * 192);
  build_cfin<<<blocks((size_t)N * 32), 256, 0, stream>>>(ncf, cfin, (size_t)N * 32);
  gemm(bufA, nf_w1t, nf_b1p, bufB, 192, 192, 192, 0.01f);   // t1
  gemm(bufB, nf_w2t, nf_b2,  bufA, 192, 192, 192, 0.01f);   // x
  zero_u16<<<blocks((size_t)N * 32), 256, 0, stream>>>(cf1, (size_t)N * 32);
  gemm(cfin, cf_w1t, cf_b1p, cf1, 32, 16, 32, 0.01f);       // cf hidden (K-padded)
  gemm(cf1,  cf_w2t, cf_b2,  cfb, 32, 64, 64, 0.01f);       // cf

  // ---- pre-net ----
  concat256<<<blocks((size_t)N * 256), 256, 0, stream>>>(cfb, bufA, bufY, 256, 0, (size_t)N * 256);
  gemm(bufY, pre_w1t, pre_b1, bufB, 256, 192, 192, 0.2f);
  gemm(bufB, pre_w2t, pre_b2, bufH, 192, 192, 192, 0.2f);   // h

  // ---- 2 x (SAGE max-aggr + MLP + residual) ----
  const u32 NEG_INF_KEY = 0x007FFFFFu;  // fkey(-inf)
  for (int l = 0; l < 2; ++l) {
    concat256<<<blocks((size_t)N * 256), 256, 0, stream>>>(cfb, bufH, zbuf, 512, 256, (size_t)N * 256);
    fill_u32<<<blocks((size_t)N * 256), 256, 0, stream>>>(aggK, NEG_INF_KEY, (size_t)N * 256);
    scatter_max<<<blocks((size_t)E * 64), 256, 0, stream>>>(edges, zbuf, aggK, E);
    decode_agg<<<blocks((size_t)N * 256), 256, 0, stream>>>(aggK, zbuf, (size_t)N * 256);
    // y1 = [agg|y] @ [wl;wr] + bl   (fused K=512 GEMM, no activation)
    gemm(zbuf, sageT[l], sage_bl + (size_t)l * 256, bufY, 512, 256, 256, 1.0f);
    gemm(bufY, mlp1T[l], mlp_b1 + (size_t)l * 192, bufA, 256, 192, 192, 0.2f);
    gemm(bufA, mlp2T[l], mlp_b2 + (size_t)l * 192, bufB, 192, 192, 192, 0.2f);
    residual_add<<<blocks((size_t)N * 192), 256, 0, stream>>>(bufH, bufB, (size_t)N * 192);
  }

  // ---- segment pooling + fc ----
  pool_fc<<<nseg, 192, 0, stream>>>(bufH, sep, fc_w, out);
}
